// GAT_5342939316773
// MI455X (gfx1250) — matrix operand order repaired and checked by reading, not compiled, not run on hardware
//
#include <hip/hip_runtime.h>
#include <cstdint>

// ---------------------------------------------------------------------------
// 2-layer GATConv forward for MI455X (gfx1250, wave32).
// GEMMs: v_wmma_f32_16x16x32_bf16, B pre-packed to WMMA lane layout so every
// operand is fed by global_load_b128. Edge softmax uses the shift-invariant
// form (no segment_max pass). Scatter-add via agent-scope f32 atomics.
// ---------------------------------------------------------------------------

typedef __bf16 bf16_t;
typedef __bf16 v8bf  __attribute__((ext_vector_type(8)));
typedef __bf16 v16bf __attribute__((ext_vector_type(16)));
typedef float  v8f   __attribute__((ext_vector_type(8)));

#define NEG_SLOPE 0.2f

__device__ __forceinline__ void atomAddF32(float* p, float v) {
  __hip_atomic_fetch_add(p, v, __ATOMIC_RELAXED, __HIP_MEMORY_SCOPE_AGENT);
}

// ---- f32 -> bf16 conversion (A matrices) -----------------------------------
__global__ void k_cvt_bf16(const float* __restrict__ in, bf16_t* __restrict__ out,
                           long n) {
  long i = (long)blockIdx.x * blockDim.x + threadIdx.x;
  if (i < n) out[i] = (bf16_t)in[i];
}

// ---- pack B[K,N] f32 -> bf16 in WMMA lane layout ---------------------------
// Bp[((tn*kblks + kb)*32 + lane)*16 + j] = B[(kb*32 + (lane>>4)*16 + j)*N + tn*16 + (lane&15)]
// so each lane's 16 operand bf16s for one (tile, k-block) are contiguous.
__global__ void k_pack_b(const float* __restrict__ B, bf16_t* __restrict__ Bp,
                         int N, int K) {
  int kblks = K >> 5;
  long total = (long)(N >> 4) * kblks * 32 * 16;
  long idx = (long)blockIdx.x * blockDim.x + threadIdx.x;
  if (idx >= total) return;
  int j    = (int)(idx & 15);
  int lane = (int)((idx >> 4) & 31);
  long t   = idx >> 9;
  int kb   = (int)(t % kblks);
  int tn   = (int)(t / kblks);
  int k    = kb * 32 + ((lane >> 4) << 4) + j;
  int col  = tn * 16 + (lane & 15);
  Bp[idx] = (bf16_t)B[(size_t)k * N + col];
}

// ---- C[M,N] = A[M,K] * Bp; one wave per 16-row strip, TN column tiles ------
// A 16x32 bf16 per lane = two contiguous 16B chunks: arow[half*16..+7], arow[half*16+8..+7].
template <int TN>
__global__ void k_wmma_gemm(const bf16_t* __restrict__ A, const bf16_t* __restrict__ Bp,
                            float* __restrict__ C, int M, int N, int K) {
  int wid  = (int)(((long)blockIdx.x * blockDim.x + threadIdx.x) >> 5);
  int lane = threadIdx.x & 31;
  int strips = M >> 4;
  if (wid >= strips) return;               // wave-uniform: EXEC stays all-1
  int r = lane & 15, half = lane >> 4;
  int kblks = K >> 5;

  const bf16_t* arow = A + (size_t)(wid * 16 + r) * K;
  v8f acc[TN] = {};
  for (int kb = 0; kb < kblks; ++kb) {
    v8bf alo = *(const v8bf*)(arow + kb * 32 + half * 16);
    v8bf ahi = *(const v8bf*)(arow + kb * 32 + half * 16 + 8);
    v16bf a = __builtin_shufflevector(alo, ahi, 0, 1, 2, 3, 4, 5, 6, 7,
                                      8, 9, 10, 11, 12, 13, 14, 15);
#pragma unroll
    for (int tn = 0; tn < TN; ++tn) {
      const bf16_t* bp = Bp + (((size_t)tn * kblks + kb) * 32 + lane) * 16;
      v8bf blo = *(const v8bf*)(bp);
      v8bf bhi = *(const v8bf*)(bp + 8);
      v16bf b = __builtin_shufflevector(blo, bhi, 0, 1, 2, 3, 4, 5, 6, 7,
                                        8, 9, 10, 11, 12, 13, 14, 15);
      acc[tn] = __builtin_amdgcn_wmma_f32_16x16x32_bf16(false, a, false, b,
                                                        (short)0, acc[tn],
                                                        false, false);
    }
  }
#pragma unroll
  for (int tn = 0; tn < TN; ++tn) {
#pragma unroll
    for (int v = 0; v < 8; ++v) {
      int row = wid * 16 + v + (half << 3);
      C[(size_t)row * N + tn * 16 + r] = acc[tn][v];
    }
  }
}

// ---- per-(node,head) attention coefficients: as = <h, a_src>, ad = <h, a_dst>
__global__ void k_attn(const float* __restrict__ h, const float* __restrict__ aw_s,
                       const float* __restrict__ aw_d, float* __restrict__ as,
                       float* __restrict__ ad, int N, int H, int C) {
  long i = (long)blockIdx.x * blockDim.x + threadIdx.x;
  if (i >= (long)N * H) return;
  int n = (int)(i / H), hh = (int)(i % H);
  const float* row = h + (size_t)n * H * C + (size_t)hh * C;
  float s = 0.f, d = 0.f;
  for (int c = 0; c < C; ++c) {
    float v = row[c];
    s += v * aw_s[hh * C + c];
    d += v * aw_d[hh * C + c];
  }
  as[i] = s;
  ad[i] = d;
}

// ---- edge pass 1: ex = exp(leaky_relu(as[src]+ad[dst])); denom[dst] += ex
// (softmax is shift-invariant; segment_max pass omitted; logits are O(0.2))
__global__ void k_edge_softmax(const int* __restrict__ srcI, const int* __restrict__ dstI,
                               int E, int N, int H, const float* __restrict__ as,
                               const float* __restrict__ ad, float* __restrict__ ex,
                               float* __restrict__ denom) {
  long e = (long)blockIdx.x * blockDim.x + threadIdx.x;
  long total = (long)E + N;
  if (e >= total) return;
  int s = (e < E) ? srcI[e] : (int)(e - E);   // self loops appended
  int d = (e < E) ? dstI[e] : (int)(e - E);
  for (int hh = 0; hh < H; ++hh) {
    float v = as[(size_t)s * H + hh] + ad[(size_t)d * H + hh];
    v = v > 0.f ? v : NEG_SLOPE * v;
    float w = __expf(v);
    ex[(size_t)e * H + hh] = w;
    atomAddF32(&denom[(size_t)d * H + hh], w);
  }
}

// ---- edge pass 2: out[dst] += (ex/denom) * h[src]; one thread per (edge, 4ch)
__global__ void k_edge_agg(const int* __restrict__ srcI, const int* __restrict__ dstI,
                           int E, int N, int H, int C, const float* __restrict__ h,
                           const float* __restrict__ ex, const float* __restrict__ denom,
                           float* __restrict__ out) {
  int F = H * C;
  int groups = F >> 2;
  long tid = (long)blockIdx.x * blockDim.x + threadIdx.x;
  long total = ((long)E + N) * groups;
  if (tid >= total) return;
  long e = tid / groups;
  int g = (int)(tid % groups);
  int s = (e < E) ? srcI[e] : (int)(e - E);
  int d = (e < E) ? dstI[e] : (int)(e - E);
  int hh = (g << 2) / C;
  float alpha = ex[(size_t)e * H + hh] / (denom[(size_t)d * H + hh] + 1e-16f);
  const float4 hv = *(const float4*)(h + (size_t)s * F + (g << 2));
  float* op = out + (size_t)d * F + (g << 2);
  atomAddF32(op + 0, alpha * hv.x);
  atomAddF32(op + 1, alpha * hv.y);
  atomAddF32(op + 2, alpha * hv.z);
  atomAddF32(op + 3, alpha * hv.w);
}

// ---- layer-1 epilogue: elu(agg + b1) -> bf16 (input of GEMM2)
__global__ void k_elu_bias_bf16(const float* __restrict__ agg, const float* __restrict__ bias,
                                bf16_t* __restrict__ out, long n, int F) {
  long i = (long)blockIdx.x * blockDim.x + threadIdx.x;
  if (i >= n) return;
  float v = agg[i] + bias[i % F];
  v = v > 0.f ? v : (__expf(v) - 1.f);
  out[i] = (bf16_t)v;
}

// ---- layer-2 epilogue: out += b2 (H=1 so mean over heads is identity)
__global__ void k_add_bias(float* __restrict__ out, const float* __restrict__ bias,
                           long n, int F) {
  long i = (long)blockIdx.x * blockDim.x + threadIdx.x;
  if (i < n) out[i] += bias[i % F];
}

// ---------------------------------------------------------------------------
extern "C" void kernel_launch(void* const* d_in, const int* in_sizes, int n_in,
                              void* d_out, int out_size, void* d_ws, size_t ws_size,
                              hipStream_t stream) {
  (void)n_in; (void)out_size; (void)ws_size;
  const float* x    = (const float*)d_in[0];
  const int*   ei   = (const int*)d_in[1];   // [2,E] (JAX default x64-off -> int32)
  const float* W1   = (const float*)d_in[2];
  const float* aS1  = (const float*)d_in[3];
  const float* aD1  = (const float*)d_in[4];
  const float* b1   = (const float*)d_in[5];
  const float* W2   = (const float*)d_in[6];
  const float* aS2  = (const float*)d_in[7];
  const float* aD2  = (const float*)d_in[8];
  const float* b2   = (const float*)d_in[9];

  const int FIN = 128, HID = 128, OUT = 64, H1 = 4, C1 = 32;
  const int N = in_sizes[0] / FIN;
  const int E = in_sizes[1] / 2;
  const int* srcI = ei;
  const int* dstI = ei + E;

  // ---- workspace layout (regions reused across layers) ----
  char* ws = (char*)d_ws;
  size_t off = 0;
  auto take = [&](size_t bytes) { size_t r = off; off += (bytes + 255) & ~(size_t)255; return r; };
  bf16_t* xb   = (bf16_t*)(ws + take((size_t)N * FIN * 2));      // later reused as h1b
  bf16_t* W1p  = (bf16_t*)(ws + take((size_t)FIN * HID * 2));    // packed
  bf16_t* W2p  = (bf16_t*)(ws + take((size_t)HID * OUT * 2));    // packed
  float*  h1   = (float*)(ws + take((size_t)N * HID * 4));       // later reused as h2
  float*  as1  = (float*)(ws + take((size_t)N * H1 * 4));        // later as2
  float*  ad1  = (float*)(ws + take((size_t)N * H1 * 4));        // later ad2
  float*  ex   = (float*)(ws + take((size_t)((size_t)E + N) * H1 * 4)); // both layers
  float*  den  = (float*)(ws + take((size_t)N * H1 * 4));        // both layers
  float*  agg1 = (float*)(ws + take((size_t)N * HID * 4));
  bf16_t* h1b  = xb;   // x dead after GEMM1
  float*  h2   = h1;   // h1 dead after layer-1 aggregation
  float*  outF = (float*)d_out;

  dim3 B(256);
  auto nb = [](long n) { return dim3((unsigned)((n + 255) / 256)); };

  // ===== precision conversion + B packing (once) =====
  { long n = (long)N * FIN;  k_cvt_bf16<<<nb(n), B, 0, stream>>>(x, xb, n); }
  { long n = (long)FIN * HID; k_pack_b<<<nb(n), B, 0, stream>>>(W1, W1p, HID, FIN); }
  { long n = (long)HID * OUT; k_pack_b<<<nb(n), B, 0, stream>>>(W2, W2p, OUT, HID); }

  // ===== layer 1 =====
  { long th = ((long)N / 16) * 32;
    k_wmma_gemm<8><<<nb(th), B, 0, stream>>>(xb, W1p, h1, N, HID, FIN); }
  { long n = (long)N * H1;
    k_attn<<<nb(n), B, 0, stream>>>(h1, aS1, aD1, as1, ad1, N, H1, C1); }
  hipMemsetAsync(den, 0, (size_t)N * H1 * 4, stream);
  { long n = (long)E + N;
    k_edge_softmax<<<nb(n), B, 0, stream>>>(srcI, dstI, E, N, H1, as1, ad1, ex, den); }
  hipMemsetAsync(agg1, 0, (size_t)N * HID * 4, stream);
  { long n = ((long)E + N) * (HID / 4);
    k_edge_agg<<<nb(n), B, 0, stream>>>(srcI, dstI, E, N, H1, C1, h1, ex, den, agg1); }
  { long n = (long)N * HID;
    k_elu_bias_bf16<<<nb(n), B, 0, stream>>>(agg1, b1, h1b, n, HID); }

  // ===== layer 2 (H=1) =====
  { long th = ((long)N / 16) * 32;
    k_wmma_gemm<4><<<nb(th), B, 0, stream>>>(h1b, W2p, h2, N, OUT, HID); }
  { long n = (long)N;
    k_attn<<<nb(n), B, 0, stream>>>(h2, aS2, aD2, as1, ad1, N, 1, OUT); }
  hipMemsetAsync(den, 0, (size_t)N * 4, stream);
  { long n = (long)E + N;
    k_edge_softmax<<<nb(n), B, 0, stream>>>(srcI, dstI, E, N, 1, as1, ad1, ex, den); }
  hipMemsetAsync(d_out, 0, (size_t)N * OUT * 4, stream);
  { long n = ((long)E + N) * (OUT / 4);
    k_edge_agg<<<nb(n), B, 0, stream>>>(srcI, dstI, E, N, 1, OUT, h2, ex, den, outF); }
  { long n = (long)N * OUT;
    k_add_bias<<<nb(n), B, 0, stream>>>(outF, b2, n, OUT); }
}
